// SLALinearCrossExtractor_47270410060108
// MI455X (gfx1250) — matrix operand order repaired and checked by reading, not compiled, not run on hardware
//
#include <hip/hip_runtime.h>
#include <hip/hip_bf16.h>

// ---------------------------------------------------------------------------
// Problem constants (compiled in; reference passes H0=96, W0=96)
// ---------------------------------------------------------------------------
#define BB   4
#define CC   256
#define NHD  8
#define HDD  32
#define HIDD 64
#define LQC  48384            // 4*96*96 + 96*96 + 48*48
#define LKC  9216             // 96*96
#define MQ   (BB * LQC)       // 193536  (multiple of 128)
#define MK   (BB * LKC)       // 36864   (multiple of 128)

typedef __attribute__((ext_vector_type(16))) __bf16 v16bf;
typedef __attribute__((ext_vector_type(8)))  float  v8f;

union Frag16 { uint4 u[2]; v16bf v; };

typedef __hip_bfloat16 bf16_t;

// ---------------------------------------------------------------------------
// CDNA5 async global->LDS copy (ASYNCcnt-tracked, no VGPR round trip).
// vdst = 32-bit LDS byte address (low 32 bits of the generic pointer),
// vaddr = 64-bit global address, saddr = off (GV mode).
// ---------------------------------------------------------------------------
__device__ __forceinline__ void async_copy16(unsigned lds_addr, const void* gaddr) {
  asm volatile("global_load_async_to_lds_b128 %0, %1, off"
               :: "v"(lds_addr), "v"(gaddr) : "memory");
}
__device__ __forceinline__ void wait_async_le3() {
  asm volatile("s_wait_asynccnt 3" ::: "memory");
}
__device__ __forceinline__ void wait_async_le0() {
  asm volatile("s_wait_asynccnt 0" ::: "memory");
}
__device__ __forceinline__ unsigned lds_off(const void* p) {
  return (unsigned)(uintptr_t)p;   // generic LDS ptr: low 32 bits = LDS offset
}

// ---------------------------------------------------------------------------
// f32 -> bf16 elementwise cast (for weights)
// ---------------------------------------------------------------------------
__global__ void cast_bf16_kernel(const float* __restrict__ in, bf16_t* __restrict__ out, long n) {
  long i = (long)blockIdx.x * 256 + threadIdx.x;
  if (i < n) out[i] = __float2bfloat16(in[i]);
}

// ---------------------------------------------------------------------------
// LayerNorm over 256 features, one wave per row, cast to bf16
// ---------------------------------------------------------------------------
__global__ void ln_cast_kernel(const float* __restrict__ x, const float* __restrict__ g,
                               const float* __restrict__ b, bf16_t* __restrict__ out,
                               long numRows) {
  int lane = threadIdx.x & 31;
  long row = (long)blockIdx.x * 8 + (threadIdx.x >> 5);
  if (row >= numRows) return;
  const float* xr = x + row * CC;
  int base = lane * 8;
  float vals[8];
  float s = 0.f, s2 = 0.f;
#pragma unroll
  for (int i = 0; i < 8; ++i) {
    float v = xr[base + i];
    vals[i] = v; s += v; s2 += v * v;
  }
#pragma unroll
  for (int off = 16; off; off >>= 1) {
    s  += __shfl_xor(s,  off);
    s2 += __shfl_xor(s2, off);
  }
  float m    = s  * (1.f / CC);
  float var  = s2 * (1.f / CC) - m * m;   // biased variance (torch LayerNorm)
  float rstd = rsqrtf(var + 1e-6f);
  bf16_t* orow = out + row * CC;
#pragma unroll
  for (int i = 0; i < 8; ++i) {
    float y = (vals[i] - m) * rstd * g[base + i] + b[base + i];
    orow[base + i] = __float2bfloat16(y);
  }
}

// ---------------------------------------------------------------------------
// WMMA GEMM:  out[M,N] (f32) = A[M,K](bf16) @ W[N,K](bf16)^T + bias[N] (+ residual)
// Block = 256 threads (8 waves), block tile 128x64, wave tile 32x32
// (four v_wmma_f32_16x16x32_bf16 per K-step, every fragment reused twice).
// Global->LDS staging uses CDNA5 async copies with double buffering so the
// next tile streams in while the current one is multiplied.
// M%128==0, N%64==0, K%32==0.
// ---------------------------------------------------------------------------
__global__ __launch_bounds__(256)
void gemm_bias_res_kernel(const bf16_t* __restrict__ A, const bf16_t* __restrict__ Wt,
                          const float* __restrict__ bias, const float* __restrict__ residual,
                          float* __restrict__ out, int M, int N, int K) {
  __shared__ __bf16 As[2][128][40];   // 40-elem stride: 80B rows, 16B aligned
  __shared__ __bf16 Ws[2][64][40];

  const int t    = threadIdx.x;
  const int lane = t & 31;
  const int w    = t >> 5;
  const int wm   = w & 3;    // 4 row groups of 32
  const int wn   = w >> 2;   // 2 col groups of 32

  const int rowBase = blockIdx.x * 128;
  const int colBase = blockIdx.y * 64;

  // fragment addressing (ISA 7.12.2 bf16 layouts)
  const int am = lane & 15;             // A: row within 16
  const int ak = (lane >> 4) * 8;       // A: K-half select {0,8}
  const int bn = lane & 15;             // B: column within 16
  const int bk = (lane >> 4) * 8;       // B: K-half select {0,8}

  v8f acc00 = {}, acc01 = {}, acc10 = {}, acc11 = {};

  // stage one 128x32 A tile + 64x32 W tile into LDS buffer `buf` (async)
  auto stage = [&](int buf, int kt) {
    // A: 512 16B-chunks, 2 per thread
#pragma unroll
    for (int i = 0; i < 2; ++i) {
      int c = t + 256 * i;
      int r = c >> 2, chk = c & 3;
      async_copy16(lds_off(&As[buf][r][chk * 8]),
                   A + (size_t)(rowBase + r) * K + kt + chk * 8);
    }
    // W: 256 16B-chunks, 1 per thread
    {
      int r = t >> 2, chk = t & 3;
      async_copy16(lds_off(&Ws[buf][r][chk * 8]),
                   Wt + (size_t)(colBase + r) * K + kt + chk * 8);
    }
  };

  const int nk = K >> 5;
  stage(0, 0);

  for (int it = 0; it < nk; ++it) {
    if (it + 1 < nk) {
      stage((it + 1) & 1, (it + 1) * 32);   // prefetch next tile (3 async ops/wave)
      wait_async_le3();                      // previous tile landed (in-order)
    } else {
      wait_async_le0();
    }
    __syncthreads();                         // tile `it` visible to all waves

    const __bf16 (*Ab)[40] = As[it & 1];
    const __bf16 (*Wb)[40] = Ws[it & 1];
    Frag16 a0, a1, b0, b1;
    a0.u[0] = *reinterpret_cast<const uint4*>(&Ab[wm * 32 + am][ak]);
    a0.u[1] = *reinterpret_cast<const uint4*>(&Ab[wm * 32 + am][16 + ak]);
    a1.u[0] = *reinterpret_cast<const uint4*>(&Ab[wm * 32 + 16 + am][ak]);
    a1.u[1] = *reinterpret_cast<const uint4*>(&Ab[wm * 32 + 16 + am][16 + ak]);
    b0.u[0] = *reinterpret_cast<const uint4*>(&Wb[wn * 32 + bn][bk]);
    b0.u[1] = *reinterpret_cast<const uint4*>(&Wb[wn * 32 + bn][bk + 16]);
    b1.u[0] = *reinterpret_cast<const uint4*>(&Wb[wn * 32 + 16 + bn][bk]);
    b1.u[1] = *reinterpret_cast<const uint4*>(&Wb[wn * 32 + 16 + bn][bk + 16]);

    acc00 = __builtin_amdgcn_wmma_f32_16x16x32_bf16(false, a0.v, false, b0.v,
                                                    (short)0, acc00, false, false);
    acc01 = __builtin_amdgcn_wmma_f32_16x16x32_bf16(false, a0.v, false, b1.v,
                                                    (short)0, acc01, false, false);
    acc10 = __builtin_amdgcn_wmma_f32_16x16x32_bf16(false, a1.v, false, b0.v,
                                                    (short)0, acc10, false, false);
    acc11 = __builtin_amdgcn_wmma_f32_16x16x32_bf16(false, a1.v, false, b1.v,
                                                    (short)0, acc11, false, false);
    __syncthreads();                         // done reading buf before it is re-staged
  }

  // epilogue: C layout — col = lane&15, row = r + (lane>>4)*8
  const int cn0 = colBase + wn * 32 + (lane & 15);
  const int cn1 = cn0 + 16;
  const int rb  = rowBase + wm * 32 + ((lane >> 4) * 8);
  const float bia0 = bias[cn0];
  const float bia1 = bias[cn1];
#pragma unroll
  for (int r = 0; r < 8; ++r) {
    const size_t row0 = (size_t)(rb + r);
    const size_t row1 = row0 + 16;
    float v00 = acc00[r] + bia0;
    float v01 = acc01[r] + bia1;
    float v10 = acc10[r] + bia0;
    float v11 = acc11[r] + bia1;
    if (residual) {
      v00 += residual[row0 * N + cn0];
      v01 += residual[row0 * N + cn1];
      v10 += residual[row1 * N + cn0];
      v11 += residual[row1 * N + cn1];
    }
    out[row0 * N + cn0] = v00;
    out[row0 * N + cn1] = v01;
    out[row1 * N + cn0] = v10;
    out[row1 * N + cn1] = v11;
  }
}

// ---------------------------------------------------------------------------
// Softmax over groups of 32 (per-head feature map), one wave per group, in place
// ---------------------------------------------------------------------------
__global__ void softmax32_kernel(float* __restrict__ buf, long numGroups,
                                 int groupsPerRow, int rowStride) {
  int lane = threadIdx.x & 31;
  long gid = (long)blockIdx.x * 8 + (threadIdx.x >> 5);
  if (gid >= numGroups) return;
  long row = gid / groupsPerRow;
  int  h   = (int)(gid % groupsPerRow);
  float* p = buf + row * (long)rowStride + h * 32 + lane;
  float v = *p;
  float mx = v;
#pragma unroll
  for (int off = 16; off; off >>= 1) mx = fmaxf(mx, __shfl_xor(mx, off));
  float e = expf(v - mx);
  float s = e;
#pragma unroll
  for (int off = 16; off; off >>= 1) s += __shfl_xor(s, off);
  *p = e / s;
}

// ---------------------------------------------------------------------------
// kv_state[b,h,d,e] = sum_lk kf[b,h,lk,d] * v[b,h,lk,e];  ksum[b,h,d] = sum_lk kf
// kv layout: [B*LK, 512] = per row {k: 8 heads x 32, v: 8 heads x 32}
// One block per (b,h), 1024 threads (one per (d,e) output), LDS-tiled over lk.
// ---------------------------------------------------------------------------
__global__ __launch_bounds__(1024)
void kv_state_kernel(const float* __restrict__ kv, float* __restrict__ kvs,
                     float* __restrict__ ksum) {
  __shared__ float kf_s[32][33];
  __shared__ float v_s[32][33];
  const int bh = blockIdx.x;
  const int b  = bh >> 3;
  const int h  = bh & 7;
  const int t  = threadIdx.x;
  const int e  = t >> 5;
  const int d  = t & 31;
  const float* base = kv + (size_t)b * LKC * 512;
  float acc = 0.f, ks = 0.f;
  for (int lk0 = 0; lk0 < LKC; lk0 += 32) {
    const int j = t >> 5, c = t & 31;
    kf_s[j][c] = base[(size_t)(lk0 + j) * 512 +       h * 32 + c];
    v_s[j][c]  = base[(size_t)(lk0 + j) * 512 + 256 + h * 32 + c];
    __syncthreads();
#pragma unroll
    for (int j2 = 0; j2 < 32; ++j2) acc += kf_s[j2][d] * v_s[j2][e];
    if (t < 32) {
#pragma unroll
      for (int j2 = 0; j2 < 32; ++j2) ks += kf_s[j2][t];
    }
    __syncthreads();
  }
  kvs[(size_t)bh * 1024 + d * 32 + e] = acc;
  if (t < 32) ksum[bh * 32 + t] = ks;
}

// ---------------------------------------------------------------------------
// o[b,lq,h*32+e] = (qf . kv_state[:,e]) / (qf . ksum + eps), cast to bf16
// grid = (LQ/8, B*NH), block 256 (8 q-rows x 32 e)
// ---------------------------------------------------------------------------
__global__ __launch_bounds__(256)
void attn_kernel(const float* __restrict__ qf, const float* __restrict__ kvs,
                 const float* __restrict__ ksum, bf16_t* __restrict__ o) {
  __shared__ float kvs_s[32][33];
  __shared__ float ks_s[32];
  __shared__ float q_s[8][33];
  const int bh = blockIdx.y;
  const int b  = bh >> 3;
  const int h  = bh & 7;
  const long lq0 = (long)blockIdx.x * 8;
  const int t = threadIdx.x;
  for (int i = t; i < 1024; i += 256) kvs_s[i >> 5][i & 31] = kvs[(size_t)bh * 1024 + i];
  if (t < 32) ks_s[t] = ksum[bh * 32 + t];
  {
    const int qi = t >> 5, d = t & 31;
    q_s[qi][d] = qf[((size_t)b * LQC + lq0 + qi) * CC + h * 32 + d];
  }
  __syncthreads();
  const int qi = t >> 5, e = t & 31;
  float num = 0.f, z = 0.f;
#pragma unroll
  for (int d = 0; d < 32; ++d) {
    const float qv = q_s[qi][d];
    num += qv * kvs_s[d][e];
    z   += qv * ks_s[d];
  }
  const float ov = num / (z + 1e-6f);
  o[((size_t)b * LQC + lq0 + qi) * CC + h * 32 + e] = __float2bfloat16(ov);
}

// ---------------------------------------------------------------------------
// Depthwise 3x3 conv over the three spatial segments (SAME zero pad) + exact GELU
// Input h1 f32 [B*LQ, 64] channel-last; output bf16 [B*LQ, 64].
// Block = 256 threads = 4 pixels x 64 channels.
// ---------------------------------------------------------------------------
__global__ __launch_bounds__(256)
void dwconv_gelu_kernel(const float* __restrict__ h1, const float* __restrict__ wk,
                        const float* __restrict__ wb, bf16_t* __restrict__ g) {
  __shared__ float wk_s[64 * 9];
  __shared__ float wb_s[64];
  const int t = threadIdx.x;
  for (int i = t; i < 576; i += 256) wk_s[i] = wk[i];
  if (t < 64) wb_s[t] = wb[t];
  __syncthreads();
  const int c = t & 63;
  const long p = (long)blockIdx.x * 4 + (t >> 6);
  const long b = p / LQC;
  const int  r = (int)(p % LQC);
  int H, W, st;
  if (r < 36864)      { H = 192; W = 192; st = 0;     }
  else if (r < 46080) { H = 96;  W = 96;  st = 36864; }
  else                { H = 48;  W = 48;  st = 46080; }
  const int pr = r - st;
  const int y = pr / W, x = pr % W;
  const float* seg = h1 + ((size_t)b * LQC + st) * HIDD + c;
  float acc = wb_s[c];
#pragma unroll
  for (int ky = 0; ky < 3; ++ky) {
    const int yy = y + ky - 1;
    if (yy < 0 || yy >= H) continue;
#pragma unroll
    for (int kx = 0; kx < 3; ++kx) {
      const int xx = x + kx - 1;
      if (xx < 0 || xx >= W) continue;
      acc += seg[(size_t)(yy * W + xx) * HIDD] * wk_s[c * 9 + ky * 3 + kx];
    }
  }
  const float gel = 0.5f * acc * (1.0f + erff(acc * 0.70710678118654752f)); // exact GELU
  g[((size_t)b * LQC + r) * HIDD + c] = __float2bfloat16(gel);
}

// ---------------------------------------------------------------------------
// Host launcher
// ---------------------------------------------------------------------------
extern "C" void kernel_launch(void* const* d_in, const int* in_sizes, int n_in,
                              void* d_out, int out_size, void* d_ws, size_t ws_size,
                              hipStream_t stream) {
  (void)in_sizes; (void)n_in; (void)out_size; (void)ws_size;

  const float* query      = (const float*)d_in[0];
  const float* feat       = (const float*)d_in[1];
  const float* q_norm_g   = (const float*)d_in[2];
  const float* q_norm_b   = (const float*)d_in[3];
  const float* kv_norm_g  = (const float*)d_in[4];
  const float* kv_norm_b  = (const float*)d_in[5];
  const float* Wq         = (const float*)d_in[6];
  const float* bq         = (const float*)d_in[7];
  const float* Wkv        = (const float*)d_in[8];
  const float* bkv        = (const float*)d_in[9];
  const float* Wo         = (const float*)d_in[10];
  const float* bo         = (const float*)d_in[11];
  const float* ffn_norm_g = (const float*)d_in[12];
  const float* ffn_norm_b = (const float*)d_in[13];
  const float* W1         = (const float*)d_in[14];
  const float* b1         = (const float*)d_in[15];
  const float* dw_k       = (const float*)d_in[16];
  const float* dw_b       = (const float*)d_in[17];
  const float* W2         = (const float*)d_in[18];
  const float* b2         = (const float*)d_in[19];
  float* out_f32 = (float*)d_out;

  // -------- workspace carve-up (256B aligned) --------
  char* ws = (char*)d_ws;
  size_t off = 0;
  auto carve = [&](size_t bytes) -> void* {
    void* p = ws + off;
    off += (bytes + 255) & ~(size_t)255;
    return p;
  };
  bf16_t* bufA = (bf16_t*)carve((size_t)MQ * CC * 2);   // qnb, later o_bf16
  bf16_t* bufB = (bf16_t*)carve((size_t)MK * CC * 2);   // knb
  float*  bufC = (float*)carve((size_t)MQ * CC * 4);    // q/qf, later x
  float*  bufD = (float*)carve((size_t)MK * 512 * 4);   // kv (kf,v), later h1
  bf16_t* bufE = (bf16_t*)carve((size_t)MQ * CC * 2);   // xnb
  bf16_t* bufF = (bf16_t*)carve((size_t)MQ * HIDD * 2); // g (post GELU)
  float*  kvs  = (float*)carve(32 * 1024 * 4);
  float*  ksum = (float*)carve(32 * 32 * 4);
  bf16_t* WqB  = (bf16_t*)carve((size_t)CC * CC * 2);
  bf16_t* WkvB = (bf16_t*)carve((size_t)2 * CC * CC * 2);
  bf16_t* WoB  = (bf16_t*)carve((size_t)CC * CC * 2);
  bf16_t* W1B  = (bf16_t*)carve((size_t)HIDD * CC * 2);
  bf16_t* W2B  = (bf16_t*)carve((size_t)CC * HIDD * 2);

  // -------- weight casts f32 -> bf16 --------
  auto cast = [&](const float* src, bf16_t* dst, long n) {
    cast_bf16_kernel<<<(unsigned)((n + 255) / 256), 256, 0, stream>>>(src, dst, n);
  };
  cast(Wq,  WqB,  (long)CC * CC);
  cast(Wkv, WkvB, (long)2 * CC * CC);
  cast(Wo,  WoB,  (long)CC * CC);
  cast(W1,  W1B,  (long)HIDD * CC);
  cast(W2,  W2B,  (long)CC * HIDD);

  // -------- 1) LayerNorms (f32 -> bf16) --------
  ln_cast_kernel<<<MQ / 8, 256, 0, stream>>>(query, q_norm_g, q_norm_b, bufA, MQ);
  ln_cast_kernel<<<MK / 8, 256, 0, stream>>>(feat, kv_norm_g, kv_norm_b, bufB, MK);

  // -------- 2) q / kv projections (WMMA) --------
  gemm_bias_res_kernel<<<dim3(MQ / 128, CC / 64), 256, 0, stream>>>(
      bufA, WqB, bq, nullptr, bufC, MQ, CC, CC);
  gemm_bias_res_kernel<<<dim3(MK / 128, 512 / 64), 256, 0, stream>>>(
      bufB, WkvB, bkv, nullptr, bufD, MK, 512, CC);

  // -------- 3) tied softmax feature maps (in place) --------
  softmax32_kernel<<<(unsigned)((long)MQ * NHD / 8), 256, 0, stream>>>(
      bufC, (long)MQ * NHD, NHD, CC);            // qf
  softmax32_kernel<<<(unsigned)((long)MK * NHD / 8), 256, 0, stream>>>(
      bufD, (long)MK * NHD, NHD, 512);           // kf (first 256 cols of kv rows)

  // -------- 4) kv_state + ksum --------
  kv_state_kernel<<<BB * NHD, 1024, 0, stream>>>(bufD, kvs, ksum);

  // -------- 5) attention readout -> o (bf16, [B,LQ,256]) --------
  attn_kernel<<<dim3(LQC / 8, BB * NHD), 256, 0, stream>>>(bufC, kvs, ksum, bufA);

  // -------- 6) output projection + residual: x = query + o @ Wo^T + bo --------
  gemm_bias_res_kernel<<<dim3(MQ / 128, CC / 64), 256, 0, stream>>>(
      bufA, WoB, bo, query, bufC, MQ, CC, CC);   // bufC now holds x

  // -------- 7) FFN --------
  ln_cast_kernel<<<MQ / 8, 256, 0, stream>>>(bufC, ffn_norm_g, ffn_norm_b, bufE, MQ);
  gemm_bias_res_kernel<<<dim3(MQ / 128, HIDD / 64), 256, 0, stream>>>(
      bufE, W1B, b1, nullptr, bufD, MQ, HIDD, CC);          // h1 (reuses kv space)
  dwconv_gelu_kernel<<<MQ / 4, 256, 0, stream>>>(bufD, dw_k, dw_b, bufF);
  gemm_bias_res_kernel<<<dim3(MQ / 128, CC / 64), 256, 0, stream>>>(
      bufF, W2B, b2, bufC, out_f32, MQ, CC, HIDD);          // out = x + ffn
}
